// FourrierKernelAttention_27719718928627
// MI455X (gfx1250) — matrix operand order repaired and checked by reading, not compiled, not run on hardware
//
#include <hip/hip_runtime.h>
#include <hip/hip_bf16.h>

// FourrierKernelAttention for MI455X (gfx1250, wave32, WMMA).
// score = A1·B1^T + A2·B2^T via sin(a-b)=sin a cos b - cos a sin b,
// streaming masked normalization (row-sum computed as S·ones via WMMA),
// then score·V.  f16 operands, f32 accumulation, ping-pong B-tile buffers.

typedef _Float16 h8_t  __attribute__((ext_vector_type(8)));
typedef _Float16 h16_t __attribute__((ext_vector_type(16)));
typedef float    f8_t  __attribute__((ext_vector_type(8)));

#define NB 2
#define HH 8
#define LQn 512
#define LKn 512
#define DDn 64
#define NHD ((size_t)NB * HH * LQn * DDn)   // 524288 elements per ws array

// ---------------------------------------------------------------- helpers ---
__device__ __forceinline__ float softplus_f(float x) {
    return fmaxf(x, 0.0f) + log1pf(__expf(-fabsf(x)));
}

// A-fragment (16x32 f16): lane m holds row m; halves = K hi*16..hi*16+15.
__device__ __forceinline__ h16_t frag_a(const _Float16* base, int stride,
                                        int m, int hi) {
    const _Float16* p = base + (size_t)m * stride + hi * 16;
    return *(const h16_t*)(p);
}

// B-fragment (32x16 f16): lane n holds column n; halves = K hi*16..hi*16+15.
__device__ __forceinline__ h16_t frag_b(const _Float16* p) {
    return *(const h16_t*)p;
}

__device__ __forceinline__ f8_t wmma_f16(h16_t a, h16_t b, f8_t c) {
    return __builtin_amdgcn_wmma_f32_16x16x32_f16(
        /*neg_a=*/false, a, /*neg_b=*/false, b,
        /*c_mod=*/(short)0, c, /*reuse_a=*/false, /*reuse_b=*/false);
}

// B-fragments for one 32-col score tile: [b1|b2] x [subtile t] x [d-half]
struct BF { h16_t f[8]; };
// B-fragments for the O matmul: 4 d-subtiles of V^T
struct VF { h16_t f[4]; };

__device__ __forceinline__ BF load_bf(const _Float16* __restrict__ B1,
                                      const _Float16* __restrict__ B2,
                                      int k0, int n16, int hi) {
    BF r;
    #pragma unroll
    for (int t = 0; t < 2; ++t) {
        const _Float16* p1 = B1 + (size_t)(k0 + 16 * t + n16) * DDn + hi * 16;
        const _Float16* p2 = B2 + (size_t)(k0 + 16 * t + n16) * DDn + hi * 16;
        r.f[t * 2 + 0]     = frag_b(p1);
        r.f[t * 2 + 1]     = frag_b(p1 + 32);
        r.f[4 + t * 2 + 0] = frag_b(p2);
        r.f[4 + t * 2 + 1] = frag_b(p2 + 32);
    }
    return r;
}

__device__ __forceinline__ VF load_vf(const _Float16* __restrict__ VT,
                                      int k0, int n16, int hi) {
    VF r;
    #pragma unroll
    for (int t2 = 0; t2 < 4; ++t2)
        r.f[t2] = frag_b(VT + (size_t)(16 * t2 + n16) * LKn + k0 + hi * 16);
    return r;
}

// ------------------------------------------------------ precompute kernel ---
__global__ __launch_bounds__(256) void fka_pre(
    const float* __restrict__ q, const float* __restrict__ k,
    const float* __restrict__ v,
    _Float16* __restrict__ a1, _Float16* __restrict__ a2,
    _Float16* __restrict__ b1, _Float16* __restrict__ b2,
    _Float16* __restrict__ vt) {
    size_t idx = (size_t)blockIdx.x * blockDim.x + threadIdx.x;
    float qv = q[idx], kv = k[idx], vv = v[idx];
    float sq, cq, sk, ck;
    __sincosf(softplus_f(qv), &sq, &cq);
    __sincosf(softplus_f(kv), &sk, &ck);
    a1[idx] = (_Float16)(qv * sq);
    a2[idx] = (_Float16)(-qv * cq);
    b1[idx] = (_Float16)(kv * ck);
    b2[idx] = (_Float16)(kv * sk);
    unsigned d  = (unsigned)idx & (DDn - 1);
    size_t rest = idx >> 6;
    unsigned l  = (unsigned)rest & (LKn - 1);
    size_t nh   = rest >> 9;
    vt[(nh * DDn + d) * LKn + l] = (_Float16)vv;   // V transposed per head
}

// ---------------------------------------------------------- one tile body ---
// Consumes preloaded B-fragments: 8 score WMMAs, mask, LDS transpose,
// 1 row-sum WMMA, 4 O WMMAs.
template <bool CAUSAL>
__device__ __forceinline__ void fka_tile(
    int k0, const BF& bf, int qb, int off, int n16, int hi,
    const _Float16* __restrict__ VT, const unsigned char* __restrict__ pm,
    _Float16* __restrict__ sl,
    const h16_t (&fa1)[2], const h16_t (&fa2)[2], const h16_t& onesf,
    f8_t (&acc)[4], f8_t& accS) {
    VF vf = load_vf(VT, k0, n16, hi);         // latency hidden under WMMAs
    const bool pad0 = pm[k0 + n16] != 0;
    const bool pad1 = pm[k0 + 16 + n16] != 0;

    f8_t S[2];
    #pragma unroll
    for (int t = 0; t < 2; ++t) {
        f8_t c = {};
        c = wmma_f16(fa2[1], bf.f[4 + t * 2 + 1], c);
        c = wmma_f16(fa2[0], bf.f[4 + t * 2 + 0], c);
        c = wmma_f16(fa1[1], bf.f[t * 2 + 1],     c);
        c = wmma_f16(fa1[0], bf.f[t * 2 + 0],     c);
        S[t] = c;
    }
    #pragma unroll
    for (int t = 0; t < 2; ++t) {
        const int  kg   = k0 + 16 * t + n16;
        const bool padv = t ? pad1 : pad0;
        #pragma unroll
        for (int r = 0; r < 8; ++r) {
            float sv  = S[t][r];
            bool keep = !padv;
            if (CAUSAL) keep = keep && (kg <= qb + r + 8 * hi + off);
            sv = keep ? sv : 0.0f;
            sl[(r + 8 * hi) * 32 + 16 * t + n16] = (_Float16)sv;
        }
    }
    asm volatile("s_wait_dscnt 0" ::: "memory");  // RAW: stores -> frag reload
    h16_t fs = frag_a(sl, 32, n16, hi);           // masked S, A-layout
    accS = wmma_f16(fs, onesf, accS);             // row sums = S · ones
    #pragma unroll
    for (int t2 = 0; t2 < 4; ++t2)
        acc[t2] = wmma_f16(fs, vf.f[t2], acc[t2]);
    // WAR (next tile's LDS stores) is safe: same-wave LDS ops are in-order
    // and the aliasing sl accesses cannot be reordered by the compiler.
}

// ------------------------------------------- k-span, ping-pong unroll by 2 ---
template <bool CAUSAL>
__device__ __forceinline__ void fka_span(
    int k0beg, int k0end, int qb, int off, int n16, int hi,
    const _Float16* __restrict__ B1, const _Float16* __restrict__ B2,
    const _Float16* __restrict__ VT, const unsigned char* __restrict__ pm,
    _Float16* __restrict__ sl,
    const h16_t (&fa1)[2], const h16_t (&fa2)[2], const h16_t& onesf,
    f8_t (&acc)[4], f8_t& accS) {
    if (k0beg >= k0end) return;
    BF b0 = load_bf(B1, B2, k0beg, n16, hi);      // prologue
    int k0 = k0beg;
    for (; k0 + 32 < k0end; k0 += 64) {
        BF b1 = load_bf(B1, B2, k0 + 32, n16, hi);       // prefetch tile+1
        fka_tile<CAUSAL>(k0, b0, qb, off, n16, hi, VT, pm, sl,
                         fa1, fa2, onesf, acc, accS);
        if (k0 + 64 < k0end)
            b0 = load_bf(B1, B2, k0 + 64, n16, hi);      // prefetch tile+2
        fka_tile<CAUSAL>(k0 + 32, b1, qb, off, n16, hi, VT, pm, sl,
                         fa1, fa2, onesf, acc, accS);
    }
    if (k0 < k0end)                                       // odd tail tile
        fka_tile<CAUSAL>(k0, b0, qb, off, n16, hi, VT, pm, sl,
                         fa1, fa2, onesf, acc, accS);
}

// ------------------------------------------------------- attention kernel ---
// grid = (N*H, LQ/64); block = 128 (4 waves); each wave owns 16 q rows.
__global__ __launch_bounds__(128) void fka_attn(
    const _Float16* __restrict__ a1, const _Float16* __restrict__ a2,
    const _Float16* __restrict__ b1, const _Float16* __restrict__ b2,
    const _Float16* __restrict__ vt, const unsigned char* __restrict__ pmask,
    const int* __restrict__ mfp, const int* __restrict__ offp,
    float* __restrict__ out) {
    __shared__ _Float16 slds[4][16 * 32];   // per-wave 16x32 f16 transpose tile

    const int lane = threadIdx.x & 31;
    const int wave = threadIdx.x >> 5;
    const int n16  = lane & 15;
    const int hi   = lane >> 4;
    const int nh   = blockIdx.x;
    const int nidx = nh >> 3;                 // batch index (H == 8)
    const int qb   = blockIdx.y * 64 + wave * 16;
    const int mf   = mfp[0];
    const int off  = offp[0];

    const _Float16* A1 = a1 + (size_t)nh * LQn * DDn + (size_t)qb * DDn;
    const _Float16* A2 = a2 + (size_t)nh * LQn * DDn + (size_t)qb * DDn;
    const _Float16* B1 = b1 + (size_t)nh * LKn * DDn;
    const _Float16* B2 = b2 + (size_t)nh * LKn * DDn;
    const _Float16* VT = vt + (size_t)nh * DDn * LKn;
    const unsigned char* pm = pmask + (size_t)nidx * LKn;

    // q-side fragments, loaded once (d 0..31 and 32..63)
    h16_t fa1[2], fa2[2];
    fa1[0] = frag_a(A1,      DDn, n16, hi);
    fa1[1] = frag_a(A1 + 32, DDn, n16, hi);
    fa2[0] = frag_a(A2,      DDn, n16, hi);
    fa2[1] = frag_a(A2 + 32, DDn, n16, hi);

    // B-fragment with ones in column 0 (lane n16==0), zeros elsewhere:
    // one WMMA per tile accumulates masked row sums into column 0.
    h16_t onesf;
    {
        const _Float16 ov = (n16 == 0) ? (_Float16)1.0f : (_Float16)0.0f;
        #pragma unroll
        for (int i = 0; i < 16; ++i) onesf[i] = ov;
    }

    f8_t acc[4] = {};                        // O accumulators (d-subtiles)
    f8_t accS   = {};                        // row-sum accumulator (col 0)

    // Wave-uniform causal bounds: kend = last needed col + 1;
    // tiles with k0+31 <= qb+off need no causal masking (interior).
    int kend = LKn;
    int kint = LKn;
    if (mf) {
        int ke = qb + 16 + off;
        kend = ke < 0 ? 0 : (ke > LKn ? LKn : ke);
        int kfull = qb + off - 31;
        kint = (kfull >= 0) ? ((kfull >> 5) << 5) + 32 : 0;
        if (kint > kend) kint = kend;
    }

    _Float16* sl = slds[wave];
    fka_span<false>(0, kint, qb, off, n16, hi, B1, B2, VT, pm, sl,
                    fa1, fa2, onesf, acc, accS);      // interior: pad only
    fka_span<true>(kint, kend, qb, off, n16, hi, B1, B2, VT, pm, sl,
                   fa1, fa2, onesf, acc, accS);       // diagonal: pad+causal

    // Row sums live in column 0 == lane hi*16 of each 16-lane group.
    float scale[8];
    #pragma unroll
    for (int r = 0; r < 8; ++r) {
        float s = __shfl(accS[r], hi << 4, 32);
        scale[r] = 1.0f / (fabsf(s) + 0.001f);
    }

    float* op = out + (size_t)nh * LQn * DDn + (size_t)qb * DDn;
    #pragma unroll
    for (int t2 = 0; t2 < 4; ++t2) {
        #pragma unroll
        for (int r = 0; r < 8; ++r) {
            op[(size_t)(r + 8 * hi) * DDn + 16 * t2 + n16] =
                acc[t2][r] * scale[r];
        }
    }
}

// ------------------------------------------------------------------ launch ---
extern "C" void kernel_launch(void* const* d_in, const int* in_sizes, int n_in,
                              void* d_out, int out_size, void* d_ws,
                              size_t ws_size, hipStream_t stream) {
    const float* q = (const float*)d_in[0];
    const float* k = (const float*)d_in[1];
    const float* v = (const float*)d_in[2];
    // d_in[3] = pq, d_in[4] = pk : unused by the reference computation
    const unsigned char* pmask = (const unsigned char*)d_in[5];  // bool mask
    const int* mf  = (const int*)d_in[6];
    const int* off = (const int*)d_in[7];
    float* out = (float*)d_out;

    // workspace: 5 f16 arrays of NHD elements (5 MB total)
    _Float16* a1 = (_Float16*)d_ws;
    _Float16* a2 = a1 + NHD;
    _Float16* b1 = a2 + NHD;
    _Float16* b2 = b1 + NHD;
    _Float16* vt = b2 + NHD;

    fka_pre<<<dim3((unsigned)(NHD / 256)), 256, 0, stream>>>(
        q, k, v, a1, a2, b1, b2, vt);
    fka_attn<<<dim3(NB * HH, LQn / 64), 128, 0, stream>>>(
        a1, a2, b1, b2, vt, pmask, mf, off, out);
}
